// AttentionRnnBlock_56873956933867
// MI455X (gfx1250) — compile-verified, matrix operand order checked by reading, not run on hardware
//
#include <hip/hip_runtime.h>
#include <math.h>

typedef __attribute__((ext_vector_type(16))) _Float16 v16h;
typedef __attribute__((ext_vector_type(8)))  float    v8f;
typedef __attribute__((ext_vector_type(4)))  int      v4i;

#define T_STEPS 2048
#define H_DIM   512
#define N_BATCH 16
#define NWG     32     // one workgroup per 16-column output tile
#define NTHR    96     // 3 waves: wave0=Q, wave1=K, wave2=V

#if __has_builtin(__builtin_amdgcn_global_load_async_to_lds_b128)
#define HAVE_ASYNC_COPY 1
#else
#define HAVE_ASYNC_COPY 0
#endif

// A/B 16-bit fragment k mapping (ISA 7.12.2, 16-bit 16x32):
//   lanes 0-15 hold K {0..7} u {16..23}; lanes 16-31 hold K {8..15} u {24..31}
//   pair p (VGPR p): k_in = ((p&4)<<2) + hi*8 + ((p&3)<<1) ; elem = 2p / 2p+1

// ---------------------------------------------------------------------------
// Pack Wq/Wk/Wv (f32 row-major (H,2H)) into f16 B fragments.
// wp[((jt*3+sel)*32+kb)*32+lane][16]  ; B[k][n] = W[col0+n][k]
// ---------------------------------------------------------------------------
__global__ void pack_weights_kernel(const float* __restrict__ Wq,
                                    const float* __restrict__ Wk,
                                    const float* __restrict__ Wv,
                                    _Float16* __restrict__ wp) {
  int gid = blockIdx.x * blockDim.x + threadIdx.x;
  const int total = NWG * 3 * 32 * 32;
  if (gid >= total) return;
  int lane = gid & 31;
  int kb   = (gid >> 5) & 31;
  int rest = gid >> 10;
  int sel  = rest % 3;
  int jt   = rest / 3;
  const float* W = (sel == 0) ? Wq : (sel == 1) ? Wk : Wv;
  int n = lane & 15, hi = lane >> 4;
  int col = jt * 16 + n;
  v16h vals;
#pragma unroll
  for (int p = 0; p < 8; ++p) {
    int k_in = ((p & 4) << 2) + hi * 8 + ((p & 3) << 1);
    int k = kb * 32 + k_in;
    vals[2 * p]     = (_Float16)W[col * (2 * H_DIM) + k];
    vals[2 * p + 1] = (_Float16)W[col * (2 * H_DIM) + k + 1];
  }
  *(v16h*)(wp + (size_t)gid * 16) = vals;
}

// ---------------------------------------------------------------------------
// Pack x (N,T,H f32) into f16 A fragments: xp[((t*16+kb)*32+lane)][16]
// ---------------------------------------------------------------------------
__global__ void pack_x_kernel(const float* __restrict__ x,
                              _Float16* __restrict__ xp) {
  int gid = blockIdx.x * blockDim.x + threadIdx.x;
  const int total = T_STEPS * 16 * 32;
  if (gid >= total) return;
  int lane = gid & 31;
  int kb   = (gid >> 5) & 15;
  int t    = gid >> 9;
  int m = lane & 15, hi = lane >> 4;
  const float* row = x + ((size_t)m * T_STEPS + t) * H_DIM + kb * 32;
  v16h vals;
#pragma unroll
  for (int p = 0; p < 8; ++p) {
    int k_in = ((p & 4) << 2) + hi * 8 + ((p & 3) << 1);
    float2 v = *(const float2*)(row + k_in);
    vals[2 * p]     = (_Float16)v.x;
    vals[2 * p + 1] = (_Float16)v.y;
  }
  *(v16h*)(xp + (size_t)gid * 16) = vals;
}

// ---------------------------------------------------------------------------
// Init: zero sync counter; scatter initial hidden into h fragment buffer 0
// (bijection: 8192 elements <-> 8192 fragment slots); zero buffer 1.
// ---------------------------------------------------------------------------
__global__ void init_state_kernel(const float* __restrict__ hidden,
                                  _Float16* __restrict__ hfrag,
                                  unsigned* __restrict__ counter) {
  int gid = blockIdx.x * blockDim.x + threadIdx.x;
  if (gid == 0) *counter = 0u;
  if (gid < N_BATCH * H_DIM) {
    int mm = gid >> 9;        // batch row
    int kh = gid & 511;       // hidden column = K offset within h half
    int kb2 = kh >> 5, k_in = kh & 31;
    int hh = (k_in >> 3) & 1;
    int p  = ((k_in & 16) >> 2) | ((k_in & 6) >> 1);
    int e  = 2 * p + (k_in & 1);
    hfrag[(kb2 * 32 + hh * 16 + mm) * 16 + e] = (_Float16)hidden[gid];
    hfrag[N_BATCH * H_DIM + gid] = (_Float16)0.0f;   // buffer 1
  }
}

// ---------------------------------------------------------------------------
// Persistent RNN kernel: 32 WGs x 96 thr. Weights staged to LDS (async copy),
// A fragments streamed from prepacked f16, h exchanged cross-WG as f16
// fragments (double buffered), own-column f32 h kept in LDS for the gate.
// ---------------------------------------------------------------------------
__global__ void __launch_bounds__(NTHR)
rnn_main_kernel(const _Float16* __restrict__ xp,
                const float* __restrict__ hidden,
                const float* __restrict__ bq,
                const float* __restrict__ bk,
                const float* __restrict__ bv,
                const _Float16* __restrict__ wp,
                _Float16* __restrict__ hfrag,   // 2 x (16*32*16) f16
                unsigned* __restrict__ counter,
                float* __restrict__ out) {      // [16*2048*512] ++ [16*512]
  __shared__ _Float16 wlds[3 * 32 * 32 * 16];   // 96 KiB weight fragments
  __shared__ float qkv[3][N_BATCH * 16];
  __shared__ float htile[N_BATCH * 16];

  const int tid  = threadIdx.x;
  const int wave = tid >> 5;          // 0=Q 1=K 2=V
  const int lane = tid & 31;
  const int l15  = lane & 15;
  const int hi   = lane >> 4;
  const int jt   = blockIdx.x;
  const int col0 = jt * 16;

  // ---- stage this WG's 96 KiB weight slice into LDS ----
  {
    const uint4* src = (const uint4*)(wp + (size_t)jt * 3 * 32 * 32 * 16);
    uint4* dst = (uint4*)wlds;
#if HAVE_ASYNC_COPY
    for (int i = tid; i < 6144; i += NTHR)
      __builtin_amdgcn_global_load_async_to_lds_b128(
          (v4i*)(src + i), (v4i*)(dst + i), 0, 0);
#if __has_builtin(__builtin_amdgcn_s_wait_asynccnt)
    __builtin_amdgcn_s_wait_asynccnt(0);
#else
    asm volatile("s_wait_asynccnt 0" ::: "memory");
#endif
#else
    for (int i = tid; i < 6144; i += NTHR) dst[i] = src[i];
#endif
  }
  // own-column f32 h tile
  for (int idx = tid; idx < N_BATCH * 16; idx += NTHR)
    htile[idx] = hidden[(idx >> 4) * H_DIM + col0 + (idx & 15)];
  __syncthreads();

  const float* bias_ptr = (wave == 0) ? bq : (wave == 1) ? bk : bv;
  const float biasv = bias_ptr[col0 + l15];

  const v16h* __restrict__ wfrag = ((const v16h*)wlds) + wave * 32 * 32;
  const v16h* __restrict__ xpv   = (const v16h*)xp;
  const v16h* __restrict__ hfv   = (const v16h*)hfrag;

  for (int t = 0; t < T_STEPS; ++t) {
    // prefetch next step's x fragments (16 KiB region)
    {
      int tn = (t + 1 < T_STEPS) ? (t + 1) : t;
      const char* pf = (const char*)(xpv + (size_t)tn * 16 * 32);
      __builtin_prefetch(pf + tid * 128, 0, 0);
      __builtin_prefetch(pf + (NTHR + tid) * 128, 0, 0);
    }
    const v16h* xa = xpv + (size_t)t * 16 * 32;
    const v16h* ha = hfv + (size_t)(t & 1) * 16 * 32;

    v8f acc = {0.f, 0.f, 0.f, 0.f, 0.f, 0.f, 0.f, 0.f};
#pragma unroll 4
    for (int kb = 0; kb < 16; ++kb) {          // x half of cat
      v16h a = xa[kb * 32 + lane];
      v16h b = wfrag[kb * 32 + lane];
      acc = __builtin_amdgcn_wmma_f32_16x16x32_f16(
          false, a, false, b, (short)0, acc, false, false);
    }
#pragma unroll 4
    for (int kb = 0; kb < 16; ++kb) {          // h half of cat
      v16h a = ha[kb * 32 + lane];
      v16h b = wfrag[(16 + kb) * 32 + lane];
      acc = __builtin_amdgcn_wmma_f32_16x16x32_f16(
          false, a, false, b, (short)0, acc, false, false);
    }

    // C layout: lane-half hi, VGPR i -> batch row (hi*8+i), column l15
#pragma unroll
    for (int i = 0; i < 8; ++i)
      qkv[wave][(hi * 8 + i) * 16 + l15] = acc[i] + biasv;
    __syncthreads();

    // gate + emit pre-update h + scatter new h into f16 fragment buffer
    _Float16* hfn = hfrag + (size_t)((t + 1) & 1) * (16 * 32 * 16);
    for (int idx = tid; idx < N_BATCH * 16; idx += NTHR) {
      int mm = idx >> 4, nn = idx & 15;
      float qv = qkv[0][idx], kv = qkv[1][idx], vv = qkv[2][idx];
      float w  = 1.0f / (1.0f + __expf(-qv * kv));
      float hp = htile[idx];
      float hn = (1.0f - w) * hp + w * tanhf(vv);
      out[((size_t)mm * T_STEPS + t) * H_DIM + col0 + nn] = hp;
      htile[idx] = hn;                         // same thread owns idx each step
      int kh = col0 + nn;                      // K position within h half
      int kb2 = kh >> 5, k_in = kh & 31;
      int hh = (k_in >> 3) & 1;
      int p  = ((k_in & 16) >> 2) | ((k_in & 6) >> 1);
      int e  = 2 * p + (k_in & 1);
      hfn[(kb2 * 32 + hh * 16 + mm) * 16 + e] = (_Float16)hn;
    }

    // grid-wide barrier (release h fragments, acquire everyone else's)
    __threadfence();
    __syncthreads();
    if (tid == 0) {
      atomicAdd(counter, 1u);
      unsigned target = (unsigned)(t + 1) * NWG;
      while (__hip_atomic_load(counter, __ATOMIC_ACQUIRE,
                               __HIP_MEMORY_SCOPE_AGENT) < target) {
        __builtin_amdgcn_s_sleep(1);
      }
    }
    __syncthreads();
    __threadfence();
  }

  // final hidden = post-update own tile
  float* out2 = out + (size_t)N_BATCH * T_STEPS * H_DIM;
  for (int idx = tid; idx < N_BATCH * 16; idx += NTHR)
    out2[(idx >> 4) * H_DIM + col0 + (idx & 15)] = htile[idx];
}

// ---------------------------------------------------------------------------
extern "C" void kernel_launch(void* const* d_in, const int* in_sizes, int n_in,
                              void* d_out, int out_size, void* d_ws, size_t ws_size,
                              hipStream_t stream) {
  (void)in_sizes; (void)n_in; (void)out_size; (void)ws_size;
  const float* x  = (const float*)d_in[0];
  const float* h0 = (const float*)d_in[1];
  const float* Wq = (const float*)d_in[2];
  const float* bq = (const float*)d_in[3];
  const float* Wk = (const float*)d_in[4];
  const float* bk = (const float*)d_in[5];
  const float* Wv = (const float*)d_in[6];
  const float* bv = (const float*)d_in[7];
  float* out = (float*)d_out;

  char* ws = (char*)d_ws;
  _Float16* wp      = (_Float16*)ws;                         // 3 MiB weights
  _Float16* xp      = (_Float16*)(ws + 3145728);             // 32 MiB x frags
  _Float16* hfrag   = (_Float16*)(ws + 3145728 + 33554432);  // 32 KiB h frags
  unsigned* counter = (unsigned*)(ws + 3145728 + 33554432 + 32768);

  const int wtotal = NWG * 3 * 32 * 32;        // 98304
  const int xtotal = T_STEPS * 16 * 32;        // 1048576
  pack_weights_kernel<<<(wtotal + 255) / 256, 256, 0, stream>>>(Wq, Wk, Wv, wp);
  pack_x_kernel<<<(xtotal + 255) / 256, 256, 0, stream>>>(x, xp);
  init_state_kernel<<<(N_BATCH * H_DIM + 255) / 256, 256, 0, stream>>>(h0, hfrag, counter);
  rnn_main_kernel<<<NWG, NTHR, 0, stream>>>(xp, h0, bq, bk, bv, wp, hfrag,
                                            counter, out);
}
